// Attention2_89395449299253
// MI455X (gfx1250) — compile-verified, matrix operand order checked by reading, not run on hardware
//
#include <hip/hip_runtime.h>
#include <hip/hip_bf16.h>
#include <math.h>

typedef __bf16 bf16_t;
typedef __attribute__((ext_vector_type(16))) __bf16 v16bf;
typedef __attribute__((ext_vector_type(8)))  float  v8f;

// Problem constants
constexpr int C   = 512;
constexpr int T   = 256;
constexpr int VV  = 25;    // tokens
constexpr int VP  = 32;    // padded tokens
constexpr int K3C = 3 * C; // 1536
constexpr int NTHREADS = 512;  // 16 wave32s
constexpr int NWAVES   = 16;

// ---------------- LDS layout (byte offsets into dynamic smem) ----------------
// All operand rows are 16B aligned; row strides padded to break bank conflicts.
constexpr int XB_RS   = 520;               // bf16 elems/row (512 + 8 pad)
constexpr int XB_OFF  = 0;                 // 32 x 520 bf16              = 33280 B (reused as out_mid)
constexpr int Q_RS    = 72;                // bf16 elems/row (64 + 8)
constexpr int Q_OFF   = 33280;             // 8 x 32 x 72 bf16           = 36864 B (reused as attn_bf)
constexpr int K_OFF   = 70144;             // 8 x 32 x 72 bf16           = 36864 B (reused as final f32)
constexpr int VT_RS   = 40;                // bf16 elems/row (32 + 8)
constexpr int VT_OFF  = 107008;            // 8 x 64 x 40 bf16           = 40960 B
constexpr int AT_RS   = 36;                // f32 elems/row (32 + 4)
constexpr int AT_OFF  = 147968;            // 8 x 32 x 36 f32            = 36864 B
constexpr int SMEM_BYTES = 184832;
constexpr int ABF_RS  = 40;                // softmaxed attn, bf16 (reuses Q region)
constexpr int ABF_OFF = Q_OFF;
constexpr int FIN_RS  = 516;               // final f32 (reuses K+VT region: 66048 B)
constexpr int FIN_OFF = K_OFF;

// Load a v16bf A/B operand in the CDNA5 16-bit 16x32 lane striping:
// elements 0..7  -> K = base+0..7, elements 8..15 -> K = base+16..23
// (caller folds the lane-half (+8) and k-step (+32k) into `p`).
__device__ inline v16bf load_ab16(const bf16_t* p) {
    union { uint4 u[2]; v16bf v; } r;
    r.u[0] = *(const uint4*)(p);
    r.u[1] = *(const uint4*)(p + 16);
    return r.v;
}

__global__ __launch_bounds__(256)
void cvt_weights_kernel(const float* __restrict__ wq, const float* __restrict__ wp,
                        bf16_t* __restrict__ wq_bf, bf16_t* __restrict__ wp_bf) {
    const int stride = gridDim.x * blockDim.x;
    const int i0 = blockIdx.x * blockDim.x + threadIdx.x;
    for (int i = i0; i < K3C * C; i += stride) wq_bf[i] = (bf16_t)wq[i];
    for (int i = i0; i < C * C;   i += stride) wp_bf[i] = (bf16_t)wp[i];
}

__global__ __launch_bounds__(NTHREADS)
void attn_fused_kernel(const float* __restrict__ x,
                       const bf16_t* __restrict__ wq_bf,
                       const bf16_t* __restrict__ wp_bf,
                       float* __restrict__ out) {
    extern __shared__ char smem[];
    bf16_t* xb   = (bf16_t*)(smem + XB_OFF);   // 32 x 512 x-tile, also out_mid
    bf16_t* qs   = (bf16_t*)(smem + Q_OFF);
    bf16_t* ks   = (bf16_t*)(smem + K_OFF);
    bf16_t* vts  = (bf16_t*)(smem + VT_OFF);   // v stored transposed [h][d][m]
    float*  attn = (float*)(smem + AT_OFF);
    bf16_t* abf  = (bf16_t*)(smem + ABF_OFF);
    float*  fin  = (float*)(smem + FIN_OFF);

    const int tid  = threadIdx.x;
    const int wave = tid >> 5;
    const int lane = tid & 31;
    const int half = lane >> 4;   // which 16-lane half
    const int l16  = lane & 15;

    const int b     = blockIdx.x;
    const int n_idx = b >> 8;     // b / 256
    const int t_idx = b & 255;
    // x[n][cc][t][v]: element offset = base_io + cc*(T*VV) + v  (same for out)
    const long base_io = ((long)n_idx * C * T + (long)t_idx) * VV;

    // ---- P0: gather/transpose x -> xb (bf16), zero pad rows 25..31 ----------
    for (int i = tid; i < C * VV; i += NTHREADS) {        // 25 iters, uniform
        const int cc = i / VV, v = i % VV;
        xb[v * XB_RS + cc] = (bf16_t)x[base_io + (long)cc * (T * VV) + v];
    }
    for (int i = tid; i < C * (VP - VV); i += NTHREADS) { // 7 iters, uniform
        const int cc = i % C, v = VV + i / C;
        xb[v * XB_RS + cc] = (bf16_t)0.0f;
    }
    __syncthreads();

    // ---- P1: qkv = xb @ wqkv^T : 96 column tiles x 2 M-tiles, K=512 ---------
    // Each job computes BOTH 16-row tiles of one 16-col slab: shared B operand,
    // two independent accumulator chains (WMMA ILP), half the weight loads.
    for (int jt = wave; jt < 96; jt += NWAVES) {          // 6 jobs/wave, uniform
        const int j = jt * 16;
        const bf16_t* arow0 = xb + l16 * XB_RS;           // rows 0..15
        const bf16_t* arow1 = xb + (16 + l16) * XB_RS;    // rows 16..31
        const bf16_t* brow  = wq_bf + (long)(j + l16) * C;
        v8f acc0 = {}, acc1 = {};
        for (int kb = 0; kb < 16; ++kb) {
            const int k0 = kb * 32 + half * 8;
            __builtin_prefetch(brow + k0 + 64, 0, 1);     // global_prefetch next K chunk
            v16bf bm = load_ab16(brow + k0);
            v16bf a0 = load_ab16(arow0 + k0);
            v16bf a1 = load_ab16(arow1 + k0);
            acc0 = __builtin_amdgcn_wmma_f32_16x16x32_bf16(false, a0, false, bm,
                                                           (short)0, acc0, false, false);
            acc1 = __builtin_amdgcn_wmma_f32_16x16x32_bf16(false, a1, false, bm,
                                                           (short)0, acc1, false, false);
        }
        const int sel = j >> 9;          // 0=q 1=k 2=v
        const int jj  = j & 511;
        const int h   = jj >> 6;
        const int d0  = jj & 63;
        if (sel == 0) {                   // q, pre-scaled by hd^-0.5 = 1/8
            bf16_t* dst = qs + h * (32 * Q_RS);
            for (int r = 0; r < 8; ++r) {
                dst[(half * 8 + r) * Q_RS + d0 + l16]        = (bf16_t)(acc0[r] * 0.125f);
                dst[(16 + half * 8 + r) * Q_RS + d0 + l16]   = (bf16_t)(acc1[r] * 0.125f);
            }
        } else if (sel == 1) {            // k
            bf16_t* dst = ks + h * (32 * Q_RS);
            for (int r = 0; r < 8; ++r) {
                dst[(half * 8 + r) * Q_RS + d0 + l16]        = (bf16_t)acc0[r];
                dst[(16 + half * 8 + r) * Q_RS + d0 + l16]   = (bf16_t)acc1[r];
            }
        } else {                          // v -> transposed [d][m]
            bf16_t* dst = vts + h * (64 * VT_RS);
            for (int r = 0; r < 8; ++r) {
                dst[(d0 + l16) * VT_RS + (half * 8 + r)]      = (bf16_t)acc0[r];
                dst[(d0 + l16) * VT_RS + (16 + half * 8 + r)] = (bf16_t)acc1[r];
            }
        }
    }
    __syncthreads();

    // ---- P2: attn = q @ k^T per head (32x32, K=64) --------------------------
    // Shared A (q rows), two B tiles (kT columns) -> two accumulator chains.
    for (int job = wave; job < 16; job += NWAVES) {       // 1 job/wave, uniform
        const int h  = job >> 1;
        const int mt = job & 1;
        const bf16_t* arow  = qs + h * (32 * Q_RS) + (mt * 16 + l16) * Q_RS;
        const bf16_t* brow0 = ks + h * (32 * Q_RS) + l16 * Q_RS;
        const bf16_t* brow1 = ks + h * (32 * Q_RS) + (16 + l16) * Q_RS;
        v8f acc0 = {}, acc1 = {};
        for (int kb = 0; kb < 2; ++kb) {
            const int k0 = kb * 32 + half * 8;
            v16bf a  = load_ab16(arow + k0);
            v16bf b0 = load_ab16(brow0 + k0);
            v16bf b1 = load_ab16(brow1 + k0);
            acc0 = __builtin_amdgcn_wmma_f32_16x16x32_bf16(false, a, false, b0,
                                                           (short)0, acc0, false, false);
            acc1 = __builtin_amdgcn_wmma_f32_16x16x32_bf16(false, a, false, b1,
                                                           (short)0, acc1, false, false);
        }
        float* dst = attn + h * (32 * AT_RS);
        for (int r = 0; r < 8; ++r) {
            dst[(mt * 16 + half * 8 + r) * AT_RS + l16]      = acc0[r];
            dst[(mt * 16 + half * 8 + r) * AT_RS + 16 + l16] = acc1[r];
        }
    }
    __syncthreads();

    // ---- P2b: softmax over the HEADS axis (8 values per (n,m)) --------------
    for (int i = tid; i < 32 * 32; i += NTHREADS) {       // 2 iters, uniform
        const int nn = i >> 5, mm = i & 31;
        float v[8]; float mx = -3.0e38f;
        for (int h = 0; h < 8; ++h) {
            v[h] = attn[h * (32 * AT_RS) + nn * AT_RS + mm];
            mx = fmaxf(mx, v[h]);
        }
        float s = 0.f;
        for (int h = 0; h < 8; ++h) { v[h] = __expf(v[h] - mx); s += v[h]; }
        const float inv = 1.0f / s;
        for (int h = 0; h < 8; ++h)
            abf[h * (32 * ABF_RS) + nn * ABF_RS + mm] = (bf16_t)(v[h] * inv);
    }
    __syncthreads();

    // ---- P3: out_mid = attn @ v per head (32x64, K=32) -> xb region ---------
    // Shared A (attn rows), two d-tiles of vT -> two accumulator chains.
    bf16_t* omid = xb;
    for (int job = wave; job < 32; job += NWAVES) {       // 2 jobs/wave, uniform
        const int h   = job >> 2;
        const int mt  = (job >> 1) & 1;
        const int dtp = job & 1;                          // covers dt = 2*dtp, 2*dtp+1
        const bf16_t* arow  = abf + h * (32 * ABF_RS) + (mt * 16 + l16) * ABF_RS;
        const bf16_t* brow0 = vts + h * (64 * VT_RS) + ((dtp * 2 + 0) * 16 + l16) * VT_RS;
        const bf16_t* brow1 = vts + h * (64 * VT_RS) + ((dtp * 2 + 1) * 16 + l16) * VT_RS;
        const int k0 = half * 8;
        v16bf a  = load_ab16(arow + k0);
        v16bf b0 = load_ab16(brow0 + k0);
        v16bf b1 = load_ab16(brow1 + k0);
        v8f acc0 = {}, acc1 = {};
        acc0 = __builtin_amdgcn_wmma_f32_16x16x32_bf16(false, a, false, b0,
                                                       (short)0, acc0, false, false);
        acc1 = __builtin_amdgcn_wmma_f32_16x16x32_bf16(false, a, false, b1,
                                                       (short)0, acc1, false, false);
        for (int r = 0; r < 8; ++r) {
            const int m = mt * 16 + half * 8 + r;
            omid[m * XB_RS + h * 64 + (dtp * 2 + 0) * 16 + l16] = (bf16_t)acc0[r];
            omid[m * XB_RS + h * 64 + (dtp * 2 + 1) * 16 + l16] = (bf16_t)acc1[r];
        }
    }
    __syncthreads();

    // ---- P4: final = out_mid @ wproj^T (32x512, K=512) -> f32 LDS -----------
    // Shared B (weights), two M-tiles -> two accumulator chains.
    for (int jt = wave; jt < 32; jt += NWAVES) {          // 2 jobs/wave, uniform
        const bf16_t* arow0 = omid + l16 * XB_RS;
        const bf16_t* arow1 = omid + (16 + l16) * XB_RS;
        const bf16_t* brow  = wp_bf + (long)(jt * 16 + l16) * C;
        v8f acc0 = {}, acc1 = {};
        for (int kb = 0; kb < 16; ++kb) {
            const int k0 = kb * 32 + half * 8;
            __builtin_prefetch(brow + k0 + 64, 0, 1);
            v16bf bm = load_ab16(brow + k0);
            v16bf a0 = load_ab16(arow0 + k0);
            v16bf a1 = load_ab16(arow1 + k0);
            acc0 = __builtin_amdgcn_wmma_f32_16x16x32_bf16(false, a0, false, bm,
                                                           (short)0, acc0, false, false);
            acc1 = __builtin_amdgcn_wmma_f32_16x16x32_bf16(false, a1, false, bm,
                                                           (short)0, acc1, false, false);
        }
        for (int r = 0; r < 8; ++r) {
            fin[(half * 8 + r) * FIN_RS + jt * 16 + l16]      = acc0[r];
            fin[(16 + half * 8 + r) * FIN_RS + jt * 16 + l16] = acc1[r];
        }
    }
    __syncthreads();

    // ---- P5: scatter back to [n,c,t,vv] (25 contiguous floats per column) ---
    for (int i = tid; i < C * VV; i += NTHREADS) {        // 25 iters, uniform
        const int cc = i / VV, v = i % VV;
        out[base_io + (long)cc * (T * VV) + v] = fin[v * FIN_RS + cc];
    }
}

extern "C" void kernel_launch(void* const* d_in, const int* in_sizes, int n_in,
                              void* d_out, int out_size, void* d_ws, size_t ws_size,
                              hipStream_t stream) {
    const float* x  = (const float*)d_in[0];   // [16,512,256,25]
    const float* wq = (const float*)d_in[1];   // [1536,512]
    const float* wp = (const float*)d_in[2];   // [512,512]
    float* outp = (float*)d_out;

    bf16_t* wq_bf = (bf16_t*)d_ws;             // 1536*512 bf16
    bf16_t* wp_bf = wq_bf + (size_t)K3C * C;   // 512*512 bf16 (total ~2.1 MB of ws)

    cvt_weights_kernel<<<512, 256, 0, stream>>>(wq, wp, wq_bf, wp_bf);

    (void)hipFuncSetAttribute((const void*)attn_fused_kernel,
                              hipFuncAttributeMaxDynamicSharedMemorySize, SMEM_BYTES);
    attn_fused_kernel<<<4096, NTHREADS, SMEM_BYTES, stream>>>(x, wq_bf, wp_bf, outp);
}
